// AttentionGate_83769042141833
// MI455X (gfx1250) — matrix operand order repaired and checked by reading, not compiled, not run on hardware
//
#include <hip/hip_runtime.h>
#include <hip/hip_bf16.h>

// ---------------------------------------------------------------------------
// MHA forward for MI455X (gfx1250), wave32 + WMMA f32_16x16x32_f16.
// ~137 GFLOP, fully L2-resident working set (88 MB < 192 MB L2) => compute /
// L2-bandwidth bound. GEMMs are register-blocked 32x64 per wave (1.5 b128
// loads per WMMA); attention is flash-style with K/V tiles staged into LDS by
// the Tensor Data Mover (double-buffered, s_wait_tensorcnt pipelined).
// ---------------------------------------------------------------------------

typedef __attribute__((ext_vector_type(16))) _Float16 v16h;
typedef __attribute__((ext_vector_type(8)))  _Float16 v8h;
typedef __attribute__((ext_vector_type(4)))  _Float16 v4h;
typedef __attribute__((ext_vector_type(8)))  float    v8f;
typedef __attribute__((ext_vector_type(4)))  float    v4f;
typedef __attribute__((ext_vector_type(4)))  unsigned int u32x4;
typedef __attribute__((ext_vector_type(4)))  int      i32x4;
typedef __attribute__((ext_vector_type(8)))  int      i32x8;

#define DIM   1024
#define HEADS 16
#define HD    64
#define BATCH 4
#define SEQ   2048
#define NTOK  (BATCH * SEQ)   // 8192

#if __has_builtin(__builtin_amdgcn_tensor_load_to_lds) && \
    __has_builtin(__builtin_amdgcn_s_wait_tensorcnt)
#define HAVE_TDM 1
#else
#define HAVE_TDM 0
#endif

// ---- WMMA helper -----------------------------------------------------------
static __device__ inline v8f wmma_f16(v16h a, v16h b, v8f c) {
  // (neg_a, A, neg_b, B, c_mod, C, reuse_a, reuse_b)
  return __builtin_amdgcn_wmma_f32_16x16x32_f16(false, a, false, b, (short)0, c,
                                                false, false);
}

// A-fragment (16x32 f16) load, contiguous k order:
//  lanes 0-15 : row = lane,    K[0..15]  (kb=0)
//  lanes16-31 : row = lane-16, K[16..31] (kb=8)
static __device__ inline v16h load_a_frag(const _Float16* rowp, int kb) {
  return *(const v16h*)(rowp + 2 * kb);
}

#if HAVE_TDM
// ---- Tensor Data Mover: 2D tile (global -> LDS), data_size = 2 bytes -------
// D# packing per CDNA5 ISA:
//  g0: count=1 | lds_addr | global_addr[56:0] | type=2
//  g1: data_size=1(2B), tensor_dim0/1, tile_dim0/1, tensor_dim0_stride
// All args are wave-uniform -> SGPR groups. 2D tensor: remaining groups zero.
// This toolchain exposes the 6-arg builtin:
//   (u32x4 g0, i32x8 g1, i32x4, i32x4, i32x8, i32 cpol)
static __device__ inline void tdm_load_2d(unsigned lds_bytes, const void* gptr,
                                          unsigned tensor_d0, unsigned tensor_d1,
                                          unsigned tile_d0, unsigned tile_d1,
                                          unsigned long long stride0_elems) {
  const unsigned long long ga = (unsigned long long)gptr;
  u32x4 g0;
  g0[0] = 1u;                                   // count=1, user mode
  g0[1] = lds_bytes;                            // lds_addr (bytes)
  g0[2] = (unsigned)(ga & 0xffffffffu);         // global_addr[31:0]
  g0[3] = (unsigned)((ga >> 32) & 0x01ffffffu)  // global_addr[56:32]
        | (2u << 30);                           // type=2 ("image")
  i32x8 g1;
  g1[0] = (int)(1u << 16);                      // workgroup_mask=0, dsize=2B
  g1[1] = (int)((tensor_d0 & 0xffffu) << 16);   // tensor_dim0[15:0]
  g1[2] = (int)((tensor_d0 >> 16) | ((tensor_d1 & 0xffffu) << 16));
  g1[3] = (int)((tensor_d1 >> 16) | ((tile_d0 & 0xffffu) << 16));
  g1[4] = (int)(tile_d1 & 0xffffu);             // tile_dim1, tile_dim2=0
  g1[5] = (int)(unsigned)(stride0_elems & 0xffffffffu);
  g1[6] = (int)(unsigned)((stride0_elems >> 32) & 0xffffu);
  g1[7] = 0;                                    // tensor_dim1_stride unused
  const i32x4 z4 = {0, 0, 0, 0};
  const i32x8 z8 = {0, 0, 0, 0, 0, 0, 0, 0};
  __builtin_amdgcn_tensor_load_to_lds(g0, g1, z4, z4, z8, 0);
}
#endif

// ---- 1) x f32 -> f16 -------------------------------------------------------
__global__ void __launch_bounds__(256)
cvt_x_kernel(const float* __restrict__ x, _Float16* __restrict__ xh, int n) {
  int i = (blockIdx.x * 256 + threadIdx.x) * 4;
  if (i + 3 < n) {
    v4f v = *(const v4f*)(x + i);
    v4h o;
#pragma unroll
    for (int j = 0; j < 4; ++j) o[j] = (_Float16)v[j];
    *(v4h*)(xh + i) = o;
  }
}

// ---- 2) pack W (row-major [j][k] f32) into per-lane B-fragment layout ------
__global__ void __launch_bounds__(32)
pack_w_kernel(const float* __restrict__ w0, const float* __restrict__ w1,
              const float* __restrict__ w2, const float* __restrict__ w3,
              _Float16* __restrict__ dst) {
  const int lane = threadIdx.x & 31;
  const int kt = blockIdx.x;          // 0..31  (K tile of 32)
  const int jt = blockIdx.y;          // 0..63  (N tile of 16)
  const int z  = blockIdx.z;          // 0..3   (wq,wk,wv,wo)
  const float* w = (z == 0) ? w0 : (z == 1) ? w1 : (z == 2) ? w2 : w3;
  const int c = lane & 15;
  const int s = lane >> 4;
  const int j  = jt * 16 + c;
  const int k0 = kt * 32 + s * 16;    // B[k][j] = W[j][k]
  const float* src = w + (size_t)j * DIM + k0;
  v16h o;
#pragma unroll
  for (int e = 0; e < 16; ++e) o[e] = (_Float16)src[e];
  _Float16* d = dst + (size_t)z * (DIM * DIM)
                    + ((size_t)(jt * 32 + kt)) * 512 + lane * 16;
  *(v16h*)d = o;
}

// ---- 3) QKV projection GEMM ------------------------------------------------
// wave -> 32 rows x 64 cols (8 WMMA per 32-K step, B reused across 2 M tiles)
__global__ void __launch_bounds__(128)
gemm_qkv_kernel(const _Float16* __restrict__ xh, const _Float16* __restrict__ wP,
                const float* __restrict__ bq, const float* __restrict__ bk,
                const float* __restrict__ bv,
                _Float16* __restrict__ qh, _Float16* __restrict__ kh,
                _Float16* __restrict__ vT) {
  const int lane  = threadIdx.x & 31;
  const int wave  = threadIdx.x >> 5;
  const int n0    = (blockIdx.x * 4 + wave) * 32;   // token tile base (32 rows)
  const int jbase = blockIdx.y * 64;                // output-col strip base
  const int z     = blockIdx.z;                     // 0=q 1=k 2=v
  const _Float16* wp = wP + (size_t)z * (DIM * DIM);
  const float* bias = (z == 0) ? bq : (z == 1) ? bk : bv;

  const int c = lane & 15, s = lane >> 4, kb = s * 8;
  const _Float16* arow0 = xh + (size_t)(n0 + c) * DIM;
  const _Float16* arow1 = arow0 + (size_t)16 * DIM;

  v8f acc[2][4] = {};
  for (int kk = 0; kk < DIM; kk += 32) {
    v16h a0 = load_a_frag(arow0 + kk, kb);
    v16h a1 = load_a_frag(arow1 + kk, kb);
    const _Float16* bp = wp + ((size_t)(jbase >> 4) * 32 + (kk >> 5)) * 512
                            + lane * 16;
    if (kk + 32 < DIM) {
      __builtin_prefetch(arow0 + kk + 32, 0, 1);
      __builtin_prefetch(arow1 + kk + 32, 0, 1);
    }
#pragma unroll
    for (int nt = 0; nt < 4; ++nt) {
      v16h b = *(const v16h*)(bp + (size_t)nt * 32 * 512);
      acc[0][nt] = wmma_f16(a0, b, acc[0][nt]);
      acc[1][nt] = wmma_f16(a1, b, acc[1][nt]);
    }
  }

  const float scale = (z == 0) ? 0.125f : 1.0f;   // 1/sqrt(64) folded into Q
#pragma unroll
  for (int mt = 0; mt < 2; ++mt) {
#pragma unroll
    for (int nt = 0; nt < 4; ++nt) {
      const int j = jbase + nt * 16 + c;
      const float bcol = bias[j];
      const int h = j >> 6, d = j & 63;
#pragma unroll
      for (int r = 0; r < 8; ++r) {
        const int n = n0 + mt * 16 + (s ? r + 8 : r);
        const int bb = n >> 11, t = n & (SEQ - 1);
        const _Float16 hv = (_Float16)((acc[mt][nt][r] + bcol) * scale);
        const size_t bh = (size_t)bb * HEADS + h;
        if (z == 0)      qh[(bh * SEQ + t) * HD + d] = hv;
        else if (z == 1) kh[(bh * SEQ + t) * HD + d] = hv;
        else             vT[(bh * HD + d) * SEQ + t] = hv;
      }
    }
  }
}

// ---- 4) flash attention ----------------------------------------------------
// One wave per (b,h, 16-query tile). K/V tiles double-buffered in LDS via the
// Tensor Data Mover (s_wait_tensorcnt pipeline); 8 WMMA per 32-key step;
// online softmax with 16-lane-half reductions; P staged via LDS.
__global__ void __launch_bounds__(32)
attn_kernel(const _Float16* __restrict__ qh, const _Float16* __restrict__ kh,
            const _Float16* __restrict__ vT, _Float16* __restrict__ aoh) {
  __shared__ __align__(32) _Float16 Plds[16 * 32];
#if HAVE_TDM
  __shared__ __align__(32) _Float16 Kbuf[2][32 * HD];  // [key][d]
  __shared__ __align__(32) _Float16 Vbuf[2][HD * 32];  // [d][key]
#endif
  const int lane = threadIdx.x & 31;
  const int bh  = blockIdx.x;       // 0..63
  const int tq0 = blockIdx.y * 16;  // query tile base
  const int c = lane & 15, s = lane >> 4, kb = s * 8;

  // Q A-fragments (d=0..31 and d=32..63), kept in registers for whole loop
  const _Float16* qrow = qh + ((size_t)bh * SEQ + tq0 + c) * HD;
  const v16h a0 = load_a_frag(qrow, kb);
  const v16h a1 = load_a_frag(qrow + 32, kb);

  v8f o[4] = {};
  float m[8], l[8];
#pragma unroll
  for (int r = 0; r < 8; ++r) { m[r] = -3.0e38f; l[r] = 0.0f; }

#if HAVE_TDM
  const _Float16* kh_bh = kh + (size_t)bh * SEQ * HD;
  const _Float16* vT_bh = vT + (size_t)bh * HD * SEQ;
  // prime buffer 0 with the first 32-key tile
  tdm_load_2d((unsigned)(size_t)&Kbuf[0][0], kh_bh,
              /*tensor_d0=*/HD, /*tensor_d1=*/SEQ,
              /*tile_d0=*/HD, /*tile_d1=*/32, /*stride0=*/HD);
  tdm_load_2d((unsigned)(size_t)&Vbuf[0][0], vT_bh,
              /*tensor_d0=*/SEQ, /*tensor_d1=*/HD,
              /*tile_d0=*/32, /*tile_d1=*/HD, /*stride0=*/SEQ);
  int buf = 0;
#endif

  for (int tk = 0; tk < SEQ; tk += 32) {
#if HAVE_TDM
    if (tk + 32 < SEQ) {
      // issue next tile into the other buffer, then wait only for current
      tdm_load_2d((unsigned)(size_t)&Kbuf[buf ^ 1][0],
                  kh_bh + (size_t)(tk + 32) * HD, HD, SEQ, HD, 32, HD);
      tdm_load_2d((unsigned)(size_t)&Vbuf[buf ^ 1][0],
                  vT_bh + (tk + 32), SEQ, HD, 32, HD, SEQ);
      __builtin_amdgcn_s_wait_tensorcnt(2);
    } else {
      __builtin_amdgcn_s_wait_tensorcnt(0);
    }
    const _Float16* Kb = &Kbuf[buf][0];
    const _Float16* Vb = &Vbuf[buf][0];
#else
    const _Float16* Kb = kh + ((size_t)bh * SEQ + tk) * HD;   // [key][d]
    const _Float16* Vb = nullptr;
#endif

    // S0: keys +0..15, S1: keys +16..31 (K dim = head_dim, 2 chunks of 32)
    v8f s0 = {}, s1 = {};
    {
      v16h b00 = *(const v16h*)(Kb + (size_t)c * HD + s * 16);
      v16h b01 = *(const v16h*)(Kb + (size_t)c * HD + 32 + s * 16);
      s0 = wmma_f16(a0, b00, s0);
      s0 = wmma_f16(a1, b01, s0);
      v16h b10 = *(const v16h*)(Kb + (size_t)(16 + c) * HD + s * 16);
      v16h b11 = *(const v16h*)(Kb + (size_t)(16 + c) * HD + 32 + s * 16);
      s1 = wmma_f16(a0, b10, s1);
      s1 = wmma_f16(a1, b11, s1);
    }

    // online softmax: element r <-> row (s ? r+8 : r); cols live across the
    // 16 lanes of each half-wave -> xor-reduce with width 16
    float alpha[8];
#pragma unroll
    for (int r = 0; r < 8; ++r) {
      float rm = fmaxf(s0[r], s1[r]);
      for (int off = 1; off < 16; off <<= 1)
        rm = fmaxf(rm, __shfl_xor(rm, off, 16));
      const float mn = fmaxf(m[r], rm);
      alpha[r] = __expf(m[r] - mn);
      m[r] = mn;
      const float p0 = __expf(s0[r] - mn);
      const float p1 = __expf(s1[r] - mn);
      s0[r] = p0; s1[r] = p1;
      float rs = p0 + p1;
      for (int off = 1; off < 16; off <<= 1) rs += __shfl_xor(rs, off, 16);
      l[r] = l[r] * alpha[r] + rs;
    }

    // stage P (16x32 f16) through LDS: C-layout scatter, A-layout gather
    __syncthreads();
#pragma unroll
    for (int r = 0; r < 8; ++r) {
      const int R = s ? r + 8 : r;
      Plds[R * 32 + c]      = (_Float16)s0[r];
      Plds[R * 32 + 16 + c] = (_Float16)s1[r];
    }
    __syncthreads();
    const v16h pa = load_a_frag(&Plds[c * 32], kb);

    // O[nt] = alpha*O[nt] + P.V
#pragma unroll
    for (int nt = 0; nt < 4; ++nt) {
      const int d = nt * 16 + c;
#if HAVE_TDM
      const v16h bv = *(const v16h*)(Vb + (size_t)d * 32 + s * 16);
#else
      const v16h bv = *(const v16h*)(vT + ((size_t)bh * HD + d) * SEQ + tk
                                        + s * 16);
#endif
#pragma unroll
      for (int r = 0; r < 8; ++r) o[nt][r] *= alpha[r];
      o[nt] = wmma_f16(pa, bv, o[nt]);
    }
#if HAVE_TDM
    buf ^= 1;
#endif
  }

  // normalize and store to [b][t][h*64+d] (row-major over DIM)
  float inv[8];
#pragma unroll
  for (int r = 0; r < 8; ++r) inv[r] = 1.0f / l[r];
  const int b_ = bh >> 4, h = bh & 15;
#pragma unroll
  for (int nt = 0; nt < 4; ++nt) {
    const int d = nt * 16 + c;
#pragma unroll
    for (int r = 0; r < 8; ++r) {
      const int t = tq0 + (s ? r + 8 : r);
      aoh[((size_t)(b_ * SEQ + t)) * DIM + h * HD + d] =
          (_Float16)(o[nt][r] * inv[r]);
    }
  }
}

// ---- 5) output projection (32x64 per wave, same as QKV) --------------------
__global__ void __launch_bounds__(128)
gemm_o_kernel(const _Float16* __restrict__ aoh, const _Float16* __restrict__ woP,
              const float* __restrict__ bo, float* __restrict__ out) {
  const int lane  = threadIdx.x & 31;
  const int wave  = threadIdx.x >> 5;
  const int n0    = (blockIdx.x * 4 + wave) * 32;
  const int jbase = blockIdx.y * 64;
  const int c = lane & 15, s = lane >> 4, kb = s * 8;
  const _Float16* arow0 = aoh + (size_t)(n0 + c) * DIM;
  const _Float16* arow1 = arow0 + (size_t)16 * DIM;

  v8f acc[2][4] = {};
  for (int kk = 0; kk < DIM; kk += 32) {
    v16h a0 = load_a_frag(arow0 + kk, kb);
    v16h a1 = load_a_frag(arow1 + kk, kb);
    const _Float16* bp = woP + ((size_t)(jbase >> 4) * 32 + (kk >> 5)) * 512
                             + lane * 16;
    if (kk + 32 < DIM) {
      __builtin_prefetch(arow0 + kk + 32, 0, 1);
      __builtin_prefetch(arow1 + kk + 32, 0, 1);
    }
#pragma unroll
    for (int nt = 0; nt < 4; ++nt) {
      v16h b = *(const v16h*)(bp + (size_t)nt * 32 * 512);
      acc[0][nt] = wmma_f16(a0, b, acc[0][nt]);
      acc[1][nt] = wmma_f16(a1, b, acc[1][nt]);
    }
  }
#pragma unroll
  for (int mt = 0; mt < 2; ++mt) {
#pragma unroll
    for (int nt = 0; nt < 4; ++nt) {
      const int j = jbase + nt * 16 + c;
      const float bcol = bo[j];
#pragma unroll
      for (int r = 0; r < 8; ++r) {
        const int n = n0 + mt * 16 + (s ? r + 8 : r);
        out[(size_t)n * DIM + j] = acc[mt][nt][r] + bcol;
      }
    }
  }
}

// ---- launch ----------------------------------------------------------------
extern "C" void kernel_launch(void* const* d_in, const int* in_sizes, int n_in,
                              void* d_out, int out_size, void* d_ws,
                              size_t ws_size, hipStream_t stream) {
  (void)in_sizes; (void)n_in; (void)out_size; (void)ws_size;
  const float* x  = (const float*)d_in[0];
  const float* wq = (const float*)d_in[1];
  const float* bq = (const float*)d_in[2];
  const float* wk = (const float*)d_in[3];
  const float* bk = (const float*)d_in[4];
  const float* wv = (const float*)d_in[5];
  const float* bv = (const float*)d_in[6];
  const float* wo = (const float*)d_in[7];
  const float* bo = (const float*)d_in[8];
  float* out = (float*)d_out;

  char* ws = (char*)d_ws;                        // 88 MB used:
  _Float16* xh  = (_Float16*)(ws);               // 16 MB x f16
  _Float16* wP  = (_Float16*)(ws + (16u << 20)); //  8 MB packed wq/wk/wv/wo
  _Float16* qh  = (_Float16*)(ws + (24u << 20)); // 16 MB Q   [bh][t][64]
  _Float16* kh  = (_Float16*)(ws + (40u << 20)); // 16 MB K   [bh][t][64]
  _Float16* vT  = (_Float16*)(ws + (56u << 20)); // 16 MB V^T [bh][64][t]
  _Float16* aoh = (_Float16*)(ws + (72u << 20)); // 16 MB attn out [n][1024]

  cvt_x_kernel<<<NTOK * DIM / 1024, 256, 0, stream>>>(x, xh, NTOK * DIM);
  pack_w_kernel<<<dim3(32, 64, 4), 32, 0, stream>>>(wq, wk, wv, wo, wP);
  gemm_qkv_kernel<<<dim3(64, 16, 3), 128, 0, stream>>>(xh, wP, bq, bk, bv,
                                                       qh, kh, vT);
  attn_kernel<<<dim3(64, 128), 32, 0, stream>>>(qh, kh, vT, aoh);
  gemm_o_kernel<<<dim3(64, 16), 128, 0, stream>>>(
      aoh, wP + (size_t)3 * DIM * DIM, bo, out);
}
